// MGDPR_47614007443644
// MI455X (gfx1250) — matrix operand order repaired and checked by reading, not compiled, not run on hardware
//
#include <hip/hip_runtime.h>

typedef __attribute__((ext_vector_type(16))) __bf16 v16bf;
typedef __attribute__((ext_vector_type(8)))  __bf16 v8bf;
typedef __attribute__((ext_vector_type(2)))  __bf16 v2bf;
typedef __attribute__((ext_vector_type(8)))  float  v8f;

constexpr int Lc = 2, Rc = 3, Kc = 3, Nc = 2048, TAUc = 64, INTERc = 512, HIDc = 128, OUTc = 2;
constexpr int RNc = Rc * Nc;
constexpr float LRELU_SLOPE = 0.01f;
constexpr float INV_GAMMA   = 1.0f / 0.9f;

// ---------- WMMA fragment helpers (CDNA5 wave32 layouts, CDNA5 ISA) ----------

__device__ inline v16bf cat8(v8bf lo, v8bf hi) {
  return __builtin_shufflevector(lo, hi, 0, 1, 2, 3, 4, 5, 6, 7,
                                 8, 9, 10, 11, 12, 13, 14, 15);
}

__device__ inline v2bf pack2(float a, float b) {
  v2bf r; r[0] = (__bf16)a; r[1] = (__bf16)b; return r;
}

// A fragment: 16x32 bf16 tile, ROW-major in LDS; stride (halfs) must be mult of 8.
// lanes 0-15: M=lane, K runs {8h..8h+7} and {16+8h..23+8h}; lanes 16-31: M=lane-16, h=1.
// Two aligned 16-byte LDS loads per lane.
__device__ inline v16bf ldsA(const __bf16* base, int stride) {
  const int lane = threadIdx.x & 31;
  const int m = lane & 15, h = lane >> 4;
  const __bf16* p = base + m * stride + 8 * h;
  const v8bf lo = *(const v8bf*)p;
  const v8bf hi = *(const v8bf*)(p + 16);
  return cat8(lo, hi);
}

// B fragment: 32x16 bf16 (KxN) tile stored COLUMN-major in LDS ([n][k], stride mult of 8).
// lanes 0-15: N=lane, K runs {8h..8h+7} and {16+8h..23+8h} -> two aligned 16-byte loads.
__device__ inline v16bf ldsBc(const __bf16* base, int stride) {
  const int lane = threadIdx.x & 31;
  const int n = lane & 15, h = lane >> 4;
  const __bf16* p = base + n * stride + 8 * h;
  const v8bf lo = *(const v8bf*)p;
  const v8bf hi = *(const v8bf*)(p + 16);
  return cat8(lo, hi);
}

__device__ inline v8f wmma_bf16(v16bf a, v16bf b, v8f c) {
  return __builtin_amdgcn_wmma_f32_16x16x32_bf16(false, a, false, b, (short)0, c,
                                                 false, false);
}

__device__ inline v8f zero8() {
  v8f z = {0.f, 0.f, 0.f, 0.f, 0.f, 0.f, 0.f, 0.f};
  return z;
}

// ---------- Kernel 1: fused diffusion  u = lrelu(((theta.T)*a) @ h @ Wd + bd) ----------
// grid (N/64, R), block 128 (4 waves). Wave w owns rows i0..i0+15, full TAU=64 cols.
__global__ __launch_bounds__(128)
void diffusion_kernel(const float* __restrict__ h,     // [R][N][TAU]  (layer input)
                      const float* __restrict__ a,     // [R][N][N]
                      const float* __restrict__ Tm,    // [R][K][N][N] (layer slice)
                      const float* __restrict__ theta, // [R][K]       (layer slice)
                      const float* __restrict__ Wd,    // [R][TAU][TAU]
                      const float* __restrict__ bd,    // [R][TAU]
                      float* __restrict__ u)           // [R][N][TAU]
{
  __shared__ __align__(16) __bf16 hC[TAUc][40];      // h tile, column-major [f][j2]
  __shared__ __align__(16) __bf16 Atile[4][16][40];  // per-wave M tile, row-major
  __shared__ __align__(16) __bf16 Ptile[4][16][72];  // prop round-trip, row-major
  __shared__ __align__(16) __bf16 Wc[TAUc][72];      // Wd[r] column-major [g][f]

  const int r    = blockIdx.y;
  const int wave = threadIdx.x >> 5;
  const int lane = threadIdx.x & 31;
  const int i0   = blockIdx.x * 64 + wave * 16;

  const float th0 = theta[r * Kc + 0];
  const float th1 = theta[r * Kc + 1];
  const float th2 = theta[r * Kc + 2];
  const float* T0 = Tm + (size_t)(r * Kc + 0) * Nc * Nc;
  const float* T1 = Tm + (size_t)(r * Kc + 1) * Nc * Nc;
  const float* T2 = Tm + (size_t)(r * Kc + 2) * Nc * Nc;
  const float* ar = a + (size_t)r * Nc * Nc;
  const float* hr = h + (size_t)r * Nc * TAUc;
  const float* Wdr = Wd + r * TAUc * TAUc;

  // Stage Wd column-major (2048 pairs / 128 threads = 16 each), packed b32 stores.
#pragma unroll
  for (int e = 0; e < 16; ++e) {
    const int idx = threadIdx.x + e * 128;
    const int g = idx & 63, fp = idx >> 6;  // fp = f pair index 0..31
    *(v2bf*)&Wc[g][2 * fp] = pack2(Wdr[(2 * fp) * TAUc + g], Wdr[(2 * fp + 1) * TAUc + g]);
  }

  v8f acc[4];
#pragma unroll
  for (int t = 0; t < 4; ++t) acc[t] = zero8();

  // Each lane loads a 16-element contiguous row-segment of the M tile.
  const int rr_ = lane >> 1;
  const int j0_ = (lane & 1) * 16;
  const float* t0p = T0 + (size_t)(i0 + rr_) * Nc + j0_;
  const float* t1p = T1 + (size_t)(i0 + rr_) * Nc + j0_;
  const float* t2p = T2 + (size_t)(i0 + rr_) * Nc + j0_;
  const float* ap  = ar + (size_t)(i0 + rr_) * Nc + j0_;

  for (int jc = 0; jc < Nc; jc += 32) {
    // Stage h[jc:jc+32, 0:64] column-major (1024 pairs / 128 threads = 8 each).
#pragma unroll
    for (int e = 0; e < 8; ++e) {
      const int idx = threadIdx.x + e * 128;
      const int f = idx & 63, jp = idx >> 6;  // jp = j pair 0..15
      *(v2bf*)&hC[f][2 * jp] = pack2(hr[(size_t)(jc + 2 * jp) * TAUc + f],
                                     hr[(size_t)(jc + 2 * jp + 1) * TAUc + f]);
    }
    // Build this wave's M tile: (th0*T0 + th1*T1 + th2*T2) * a  -> bf16, packed stores.
#pragma unroll
    for (int e = 0; e < 16; e += 4) {
      const float4 t0 = *(const float4*)(t0p + jc + e);
      const float4 t1 = *(const float4*)(t1p + jc + e);
      const float4 t2 = *(const float4*)(t2p + jc + e);
      const float4 av = *(const float4*)(ap + jc + e);
      const float m0 = (th0 * t0.x + th1 * t1.x + th2 * t2.x) * av.x;
      const float m1 = (th0 * t0.y + th1 * t1.y + th2 * t2.y) * av.y;
      const float m2 = (th0 * t0.z + th1 * t1.z + th2 * t2.z) * av.z;
      const float m3 = (th0 * t0.w + th1 * t1.w + th2 * t2.w) * av.w;
      *(v2bf*)&Atile[wave][rr_][j0_ + e]     = pack2(m0, m1);
      *(v2bf*)&Atile[wave][rr_][j0_ + e + 2] = pack2(m2, m3);
    }
    // Prefetch next chunk of the dominant T stream (global_prefetch_b8).
    __builtin_prefetch(t0p + jc + 32, 0, 0);
    __builtin_prefetch(t1p + jc + 32, 0, 0);
    __builtin_prefetch(t2p + jc + 32, 0, 0);
    __syncthreads();

    const v16bf af = ldsA(&Atile[wave][0][0], 40);
#pragma unroll
    for (int nt = 0; nt < 4; ++nt) {
      const v16bf bfrag = ldsBc(&hC[nt * 16][0], 40);
      acc[nt] = wmma_bf16(af, bfrag, acc[nt]);
    }
    __syncthreads();
  }

  // Epilogue: u = lrelu(prop @ Wd + bd). Round-trip prop through LDS (C->A layout).
  const int half = lane >> 4, nn = lane & 15;
#pragma unroll
  for (int nt = 0; nt < 4; ++nt)
#pragma unroll
    for (int p = 0; p < 8; ++p)
      Ptile[wave][p + 8 * half][nt * 16 + nn] = (__bf16)acc[nt][p];
  __syncthreads();

  const float* bdr = bd + r * TAUc;
#pragma unroll
  for (int nt = 0; nt < 4; ++nt) {
    v8f c = zero8();
#pragma unroll
    for (int ks = 0; ks < 2; ++ks) {
      const v16bf af2 = ldsA(&Ptile[wave][0][ks * 32], 72);
      const v16bf bf2 = ldsBc(&Wc[nt * 16][ks * 32], 72);
      c = wmma_bf16(af2, bf2, c);
    }
#pragma unroll
    for (int p = 0; p < 8; ++p) {
      const int row = i0 + p + 8 * half;
      const int col = nt * 16 + nn;
      float v = c[p] + bdr[col];
      v = v >= 0.f ? v : LRELU_SLOPE * v;
      u[((size_t)r * Nc + row) * TAUc + col] = v;
    }
  }
}

// ---------- Kernel 2: h[s,i,f] = lrelu(sum_r Wf[s,r]*u[r,i,f] + bf[s]) ----------
__global__ __launch_bounds__(256)
void relmix_kernel(const float* __restrict__ u, const float* __restrict__ Wf,
                   const float* __restrict__ bfv, float* __restrict__ hout) {
  const int idx = blockIdx.x * 256 + threadIdx.x;
  if (idx >= Rc * Nc * TAUc) return;
  const int s = idx / (Nc * TAUc);
  const int rem = idx - s * (Nc * TAUc);
  float acc = bfv[s];
#pragma unroll
  for (int r = 0; r < Rc; ++r) acc += Wf[s * Rc + r] * u[(size_t)r * Nc * TAUc + rem];
  hout[idx] = acc >= 0.f ? acc : LRELU_SLOPE * acc;
}

// ---------- Kernel 3: X[t, r*N+i] = u[r,i,t] ----------
__global__ __launch_bounds__(256)
void transpose_kernel(const float* __restrict__ u, float* __restrict__ X) {
  const int idx = blockIdx.x * 256 + threadIdx.x;
  if (idx >= TAUc * RNc) return;
  const int t = idx / RNc, c = idx - t * RNc;
  X[idx] = u[(size_t)c * TAUc + t];
}

// ---------- Kernel 4: generic bf16-WMMA GEMM  C = act(A@B + bias (+ addend)) ----------
// A: [M,K] f32, B: [K,N] f32 (staged to LDS as bf16; B column-major). M%16==0, K%32==0.
// storeT: C[n*M + m] instead of C[m*N + n]. act: 0 none, 1 lrelu, 2 PReLU(*slopePtr).
__global__ __launch_bounds__(128)
void gemm_kernel(const float* __restrict__ A, long sAb,
                 const float* __restrict__ B, long sBb,
                 const float* __restrict__ bias, long sBiasb,
                 const float* __restrict__ addend, long sAddb,
                 float* __restrict__ C, long sCb,
                 int M, int Nd, int K, int act,
                 const float* __restrict__ slopePtr, int storeT) {
  const int batch = blockIdx.z;
  A += (size_t)batch * sAb;
  B += (size_t)batch * sBb;
  if (bias) bias += (size_t)batch * sBiasb;
  if (addend) addend += (size_t)batch * sAddb;
  C += (size_t)batch * sCb;

  __shared__ __align__(16) __bf16 As[16][40];  // row-major 16x32
  __shared__ __align__(16) __bf16 Bc[64][40];  // column-major [n][k] 32x64

  const int m0 = blockIdx.x * 16;
  const int n0blk = blockIdx.y * 64;
  const int wave = threadIdx.x >> 5, lane = threadIdx.x & 31;
  const int n0 = n0blk + wave * 16;

  v8f acc = zero8();
  for (int kc = 0; kc < K; kc += 32) {
    // A tile: 256 pairs / 128 threads = 2 each; contiguous float2 reads.
#pragma unroll
    for (int e = 0; e < 2; ++e) {
      const int idx = threadIdx.x + e * 128;
      const int m = idx >> 4, kp = idx & 15;
      const float2 av = *(const float2*)&A[(size_t)(m0 + m) * K + kc + 2 * kp];
      *(v2bf*)&As[m][2 * kp] = pack2(av.x, av.y);
    }
    // B tile: 1024 pairs / 128 threads = 8 each; coalesced along n, packed along k.
#pragma unroll
    for (int e = 0; e < 8; ++e) {
      const int idx = threadIdx.x + e * 128;
      const int n = idx & 63, kp = idx >> 6;  // kp 0..15
      const int gn = n0blk + n;
      const float b0 = (gn < Nd) ? B[(size_t)(kc + 2 * kp) * Nd + gn] : 0.f;
      const float b1 = (gn < Nd) ? B[(size_t)(kc + 2 * kp + 1) * Nd + gn] : 0.f;
      *(v2bf*)&Bc[n][2 * kp] = pack2(b0, b1);
    }
    __syncthreads();
    const v16bf af = ldsA(&As[0][0], 40);
    const v16bf bfrag = ldsBc(&Bc[wave * 16][0], 40);
    acc = wmma_bf16(af, bfrag, acc);
    __syncthreads();
  }

  const float slope = (act == 2) ? slopePtr[0] : LRELU_SLOPE;
  const int half = lane >> 4, nn = lane & 15;
  const int gn = n0 + nn;
  if (gn < Nd) {
#pragma unroll
    for (int p = 0; p < 8; ++p) {
      const int gm = m0 + p + 8 * half;
      float v = acc[p];
      if (bias) v += bias[gn];
      if (addend) v += addend[(size_t)gm * Nd + gn];
      if (act) v = v >= 0.f ? v : slope * v;
      if (storeT) C[(size_t)gn * M + gm] = v;
      else        C[(size_t)gm * Nd + gn] = v;
    }
  }
}

// ---------- Kernel 5: retention core  ret = ((q @ k^T) * D) @ v ----------
// q,k,v: [TAU, INTER]. D[i,j] = 1/gamma if i>j else 0. One block, 4 waves.
__global__ __launch_bounds__(128)
void retention_kernel(const float* __restrict__ q, const float* __restrict__ k,
                      const float* __restrict__ v, float* __restrict__ ret) {
  __shared__ __align__(16) __bf16 S[TAUc][72];    // masked scores (A side), row-major
  __shared__ __align__(16) __bf16 qa[TAUc][40];   // q tile, row-major 64x32
  __shared__ __align__(16) __bf16 ktC[TAUc][40];  // k^T tile, column-major [n][j]
  __shared__ __align__(16) __bf16 vC[TAUc][72];   // v tile, column-major [n][k] 64x64

  const int wave = threadIdx.x >> 5, lane = threadIdx.x & 31;
  const int half = lane >> 4, nn = lane & 15;

  v8f acc[4];
#pragma unroll
  for (int t = 0; t < 4; ++t) acc[t] = zero8();

  // Phase 1: scores = q @ k^T  (64x64, K=INTER)
  for (int kc = 0; kc < INTERc; kc += 32) {
#pragma unroll
    for (int e = 0; e < 8; ++e) {  // q[0:64, kc:kc+32], 1024 pairs
      const int idx = threadIdx.x + e * 128;
      const int rr = idx >> 4, kp = idx & 15;
      const float2 qv = *(const float2*)&q[(size_t)rr * INTERc + kc + 2 * kp];
      *(v2bf*)&qa[rr][2 * kp] = pack2(qv.x, qv.y);
    }
#pragma unroll
    for (int e = 0; e < 8; ++e) {  // ktC[n][j] = k[n, kc+j], contiguous float2 reads
      const int idx = threadIdx.x + e * 128;
      const int n = idx >> 4, jp = idx & 15;
      const float2 kv = *(const float2*)&k[(size_t)n * INTERc + kc + 2 * jp];
      *(v2bf*)&ktC[n][2 * jp] = pack2(kv.x, kv.y);
    }
    __syncthreads();
    const v16bf af = ldsA(&qa[wave * 16][0], 40);
#pragma unroll
    for (int nt = 0; nt < 4; ++nt) {
      const v16bf bfrag = ldsBc(&ktC[nt * 16][0], 40);
      acc[nt] = wmma_bf16(af, bfrag, acc[nt]);
    }
    __syncthreads();
  }
  // Decay mask + store bf16 scores to LDS (A-side, row-major).
#pragma unroll
  for (int nt = 0; nt < 4; ++nt)
#pragma unroll
    for (int p = 0; p < 8; ++p) {
      const int gi = wave * 16 + p + 8 * half;
      const int gj = nt * 16 + nn;
      const float s = (gi > gj) ? acc[nt][p] * INV_GAMMA : 0.f;
      S[gi][gj] = (__bf16)s;
    }
  __syncthreads();

  // Phase 2: ret = S @ v  (64x512, K=64), 64-col chunks.
  for (int nc = 0; nc < INTERc; nc += 64) {
#pragma unroll
    for (int e = 0; e < 16; ++e) {  // vC[n][kk] = v[kk, nc+n], 2048 pairs
      const int idx = threadIdx.x + e * 128;
      const int n = idx & 63, kp = idx >> 6;  // kp 0..31
      const float v0 = v[(size_t)(2 * kp) * INTERc + nc + n];
      const float v1 = v[(size_t)(2 * kp + 1) * INTERc + nc + n];
      *(v2bf*)&vC[n][2 * kp] = pack2(v0, v1);
    }
    __syncthreads();
#pragma unroll
    for (int nt = 0; nt < 4; ++nt) {
      v8f c = zero8();
#pragma unroll
      for (int ks = 0; ks < 2; ++ks) {
        const v16bf af2 = ldsA(&S[wave * 16][ks * 32], 72);
        const v16bf bf2 = ldsBc(&vC[nt * 16][ks * 32], 72);
        c = wmma_bf16(af2, bf2, c);
      }
#pragma unroll
      for (int p = 0; p < 8; ++p)
        ret[(size_t)(wave * 16 + p + 8 * half) * INTERc + nc + nt * 16 + nn] = c[p];
    }
    __syncthreads();
  }
}

// ---------- Host orchestration ----------
extern "C" void kernel_launch(void* const* d_in, const int* in_sizes, int n_in,
                              void* d_out, int out_size, void* d_ws, size_t ws_size,
                              hipStream_t stream) {
  const float* x  = (const float*)d_in[0];
  const float* a  = (const float*)d_in[1];
  const float* T  = (const float*)d_in[2];
  const float* th = (const float*)d_in[3];
  const float* Wd = (const float*)d_in[4];
  const float* bd = (const float*)d_in[5];
  const float* Wf = (const float*)d_in[6];
  const float* bfv = (const float*)d_in[7];
  const float* Wq = (const float*)d_in[8];
  const float* bq = (const float*)d_in[9];
  const float* Wk = (const float*)d_in[10];
  const float* bk = (const float*)d_in[11];
  const float* Wv = (const float*)d_in[12];
  const float* bv = (const float*)d_in[13];
  const float* Wr = (const float*)d_in[14];
  const float* br = (const float*)d_in[15];
  const float* Wl = (const float*)d_in[16];
  const float* bl = (const float*)d_in[17];
  const float* W1 = (const float*)d_in[18];
  const float* b1 = (const float*)d_in[19];
  const float* W2 = (const float*)d_in[20];
  const float* b2 = (const float*)d_in[21];
  const float* alpha = (const float*)d_in[22];
  float* out = (float*)d_out;

  float* ws = (float*)d_ws;
  float* hbuf = ws; ws += Rc * Nc * TAUc;
  float* ubuf = ws; ws += Rc * Nc * TAUc;
  float* Xbuf = ws; ws += TAUc * RNc;
  float* qb = ws;   ws += TAUc * INTERc;
  float* kb = ws;   ws += TAUc * INTERc;
  float* vb = ws;   ws += TAUc * INTERc;
  float* rb = ws;   ws += TAUc * INTERc;
  float* eta = ws;  ws += RNc * TAUc;
  float* hp0 = ws;  ws += Rc * Nc * TAUc;
  float* hp1 = ws;  ws += Rc * Nc * TAUc;
  float* zb = ws;   ws += RNc * HIDc;

  auto launch_gemm = [&](const float* A, long sA, const float* B, long sB,
                         const float* bias, long sBias, const float* add, long sAdd,
                         float* C, long sC, int M, int Nd, int Kd, int act,
                         const float* slope, int storeT, int batch) {
    dim3 grid(M / 16, (Nd + 63) / 64, batch);
    gemm_kernel<<<grid, 128, 0, stream>>>(A, sA, B, sB, bias, sBias, add, sAdd, C, sC,
                                          M, Nd, Kd, act, slope, storeT);
  };

  for (int l = 0; l < Lc; ++l) {
    const float* hcur = (l == 0) ? x : hbuf;
    const float* base = (l == 0) ? x : hp0;
    float* hpOut = (l == 0) ? hp0 : hp1;

    diffusion_kernel<<<dim3(Nc / 64, Rc), 128, 0, stream>>>(
        hcur, a, T + (size_t)l * Rc * Kc * Nc * Nc, th + l * Rc * Kc,
        Wd + l * Rc * TAUc * TAUc, bd + l * Rc * TAUc, ubuf);

    if (l + 1 < Lc)
      relmix_kernel<<<(Rc * Nc * TAUc + 255) / 256, 256, 0, stream>>>(
          ubuf, Wf + l * Rc * Rc, bfv + l * Rc, hbuf);

    transpose_kernel<<<(TAUc * RNc + 255) / 256, 256, 0, stream>>>(ubuf, Xbuf);

    launch_gemm(Xbuf, 0, Wq + (size_t)l * RNc * INTERc, 0, bq + l * INTERc, 0,
                nullptr, 0, qb, 0, TAUc, INTERc, RNc, 0, nullptr, 0, 1);
    launch_gemm(Xbuf, 0, Wk + (size_t)l * RNc * INTERc, 0, bk + l * INTERc, 0,
                nullptr, 0, kb, 0, TAUc, INTERc, RNc, 0, nullptr, 0, 1);
    launch_gemm(Xbuf, 0, Wv + (size_t)l * RNc * INTERc, 0, bv + l * INTERc, 0,
                nullptr, 0, vb, 0, TAUc, INTERc, RNc, 0, nullptr, 0, 1);

    retention_kernel<<<1, 128, 0, stream>>>(qb, kb, vb, rb);

    // eta stored transposed: eta[(r*N+i)*TAU + t]
    launch_gemm(rb, 0, Wr + (size_t)l * INTERc * RNc, 0, br + l * RNc, 0,
                nullptr, 0, eta, 0, TAUc, RNc, INTERc, 1, nullptr, 1, 1);

    // hp[r] = base[r] @ Wl + bl + eta[r]   (batched over R)
    launch_gemm(base, (long)Nc * TAUc, Wl + l * TAUc * TAUc, 0, bl + l * TAUc, 0,
                eta, (long)Nc * TAUc, hpOut, (long)Nc * TAUc,
                Nc, TAUc, TAUc, 0, nullptr, 0, Rc);
  }

  // z = PReLU(hp @ W1 + b1), out = z @ W2 + b2
  launch_gemm(hp1, 0, W1, 0, b1, 0, nullptr, 0, zb, 0, RNc, HIDc, TAUc, 2, alpha, 0, 1);
  launch_gemm(zb, 0, W2, 0, b2, 0, nullptr, 0, out, 0, RNc, OUTc, HIDc, 0, nullptr, 0, 1);
}
